// NeighborAttention_6296422056679
// MI455X (gfx1250) — matrix operand order repaired and checked by reading, not compiled, not run on hardware
//
#include <hip/hip_runtime.h>

// ---------------------------------------------------------------------------
// NeighborAttention for MI455X (gfx1250, wave32, WMMA).
// GEMM1/GEMM2 run on v_wmma_f32_16x16x32_bf16 with bf16 hi/lo error
// compensation (3 WMMAs per K-tile) => ~fp32 accuracy at bf16 matrix rate.
// Round 2: double-buffered B-operand LDS loads + split accumulator chains
// so WMMAs are not serialized behind s_wait_dscnt 0.
// ---------------------------------------------------------------------------

typedef __bf16 v16bf __attribute__((ext_vector_type(16)));
typedef __bf16 v8bf  __attribute__((ext_vector_type(8)));
typedef float  v8f   __attribute__((ext_vector_type(8)));
typedef float  v4f   __attribute__((ext_vector_type(4)));

#define NFEATS  128
#define KDIM    256      // 2*NFEATS
#define HIDDEN  256
#define CH      8
#define BSTRIDE 264      // LDS row stride in bf16 elements (256 + 8 pad -> bank spread)
#define HSTRIDE 33       // LDS h-staging row stride in floats

#define CONCAT16(a, b) __builtin_shufflevector(a, b, 0,1,2,3,4,5,6,7,8,9,10,11,12,13,14,15)

// order-preserving float <-> uint mapping for atomicMax-based segment max
__device__ __forceinline__ unsigned flip_f(float f) {
  unsigned b = __float_as_uint(f);
  return (b & 0x80000000u) ? ~b : (b | 0x80000000u);
}
__device__ __forceinline__ float unflip_f(unsigned k) {
  unsigned b = (k & 0x80000000u) ? (k & 0x7FFFFFFFu) : ~k;
  return __uint_as_float(b);
}

__device__ __forceinline__ void bf16_split(float f, __bf16& hi, __bf16& lo) {
  hi = (__bf16)f;
  lo = (__bf16)(f - (float)hi);
}

__device__ __forceinline__ v16bf load_b16(const __bf16* p) {
  v8bf a = *(const v8bf*)p;
  v8bf b = *(const v8bf*)(p + 8);
  return CONCAT16(a, b);
}

// ---- prep: W1^T (256x256) as bf16 hi/lo ------------------------------------
__global__ void prep_w1t(const float* __restrict__ W1,
                         __bf16* __restrict__ hi, __bf16* __restrict__ lo) {
  int idx = blockIdx.x * blockDim.x + threadIdx.x;   // 65536 threads
  int k = idx & 255, n = idx >> 8;
  float v = W1[k * HIDDEN + n];
  __bf16 h, l; bf16_split(v, h, l);
  hi[n * KDIM + k] = h;
  lo[n * KDIM + k] = l;
}

// ---- prep: W2^T padded to 16x256 as bf16 hi/lo -----------------------------
__global__ void prep_w2t(const float* __restrict__ W2,
                         __bf16* __restrict__ hi, __bf16* __restrict__ lo) {
  int idx = blockIdx.x * blockDim.x + threadIdx.x;   // 4096 threads
  int k = idx & 255, n = idx >> 8;                   // n in 0..15
  float v = (n < CH) ? W2[k * CH + n] : 0.0f;
  __bf16 h, l; bf16_split(v, h, l);
  hi[n * KDIM + k] = h;
  lo[n * KDIM + k] = l;
}

// ---- init segment buffers --------------------------------------------------
__global__ void init_seg(unsigned* __restrict__ seg_key,
                         float* __restrict__ seg_sum, int total) {
  int i = blockIdx.x * blockDim.x + threadIdx.x;
  if (i < total) {
    seg_key[i] = 0x007FFFFFu;   // flip_f(-inf)
    seg_sum[i] = 0.0f;
  }
}

// ---------------------------------------------------------------------------
// Main fused kernel: gather + GEMM1 (relu) + GEMM2, writes scores to d_out
// and atomically accumulates per-destination-node max.
// 256 threads = 8 waves; each wave owns one 16-edge M-tile.
// ---------------------------------------------------------------------------
__global__ __launch_bounds__(256)
void edge_mlp_kernel(const float*  __restrict__ x,
                     const int*    __restrict__ ei,
                     const __bf16* __restrict__ w1t_hi,
                     const __bf16* __restrict__ w1t_lo,
                     const __bf16* __restrict__ w2t_hi,
                     const __bf16* __restrict__ w2t_lo,
                     const float*  __restrict__ b1,
                     const float*  __restrict__ b2,
                     float*        __restrict__ score_out,
                     unsigned*     __restrict__ seg_key,
                     int E) {
  __shared__ __bf16 lds_bhi[16 * BSTRIDE];
  __shared__ __bf16 lds_blo[16 * BSTRIDE];
  __shared__ __bf16 lds_b2hi[16 * BSTRIDE];
  __shared__ __bf16 lds_b2lo[16 * BSTRIDE];
  __shared__ float  lds_h[8][16 * HSTRIDE];   // per-wave 16x32 h staging

  const int tid  = threadIdx.x;
  const int wave = tid >> 5;
  const int lane = tid & 31;
  const int g    = lane >> 4;   // K-half selector (ISA A/B lane layout)
  const int m    = lane & 15;   // A: row; B/C: column

  const int  tile  = blockIdx.x * 8 + wave;
  const bool valid = (tile * 16) < E;
  const int  e0    = valid ? (tile * 16 + m) : m;

  // ---- W2^T slab -> LDS (once per block) ----
  {
    int n  = tid >> 4;
    int kc = (tid & 15) * 16;
    *(v4f*)&lds_b2hi[n * BSTRIDE + kc]     = *(const v4f*)&w2t_hi[n * KDIM + kc];
    *(v4f*)&lds_b2hi[n * BSTRIDE + kc + 8] = *(const v4f*)&w2t_hi[n * KDIM + kc + 8];
    *(v4f*)&lds_b2lo[n * BSTRIDE + kc]     = *(const v4f*)&w2t_lo[n * KDIM + kc];
    *(v4f*)&lds_b2lo[n * BSTRIDE + kc + 8] = *(const v4f*)&w2t_lo[n * KDIM + kc + 8];
  }

  // ---- gather edge_h = concat(x[row], x[col]) into A operands (bf16 hi/lo)
  // A 16x32 in contiguous k order (matches B): e[j] = K(16g+j)
  const int r0 = ei[e0];
  const int c0 = ei[E + e0];
  const float* rowp = x + (long)r0 * NFEATS;
  const float* colp = x + (long)c0 * NFEATS;

  v16bf a_hi[8], a_lo[8];
#pragma unroll
  for (int kt = 0; kt < 8; ++kt) {
    const float* src = (kt < 4) ? rowp : colp;
    const int kb = (kt & 3) * 32 + 16 * g;
    v4f c00 = *(const v4f*)(src + kb);
    v4f c01 = *(const v4f*)(src + kb + 4);
    v4f c10 = *(const v4f*)(src + kb + 8);
    v4f c11 = *(const v4f*)(src + kb + 12);
#pragma unroll
    for (int i = 0; i < 4; ++i) {
      __bf16 h, l;
      bf16_split(c00[i], h, l); a_hi[kt][i]      = h; a_lo[kt][i]      = l;
      bf16_split(c01[i], h, l); a_hi[kt][4 + i]  = h; a_lo[kt][4 + i]  = l;
      bf16_split(c10[i], h, l); a_hi[kt][8 + i]  = h; a_lo[kt][8 + i]  = l;
      bf16_split(c11[i], h, l); a_hi[kt][12 + i] = h; a_lo[kt][12 + i] = l;
    }
  }

  v8f acc2a = {0.f, 0.f, 0.f, 0.f, 0.f, 0.f, 0.f, 0.f};
  v8f acc2b = {0.f, 0.f, 0.f, 0.f, 0.f, 0.f, 0.f, 0.f};

#pragma unroll 1
  for (int nt = 0; nt < 16; ++nt) {
    __syncthreads();
    // cooperative load of W1^T slab (16 hidden cols x 256 K) into LDS
    {
      int n  = tid >> 4;
      int kc = (tid & 15) * 16;
      const __bf16* gh = w1t_hi + (long)(16 * nt + n) * KDIM + kc;
      const __bf16* gl = w1t_lo + (long)(16 * nt + n) * KDIM + kc;
      *(v4f*)&lds_bhi[n * BSTRIDE + kc]     = *(const v4f*)gh;
      *(v4f*)&lds_bhi[n * BSTRIDE + kc + 8] = *(const v4f*)(gh + 8);
      *(v4f*)&lds_blo[n * BSTRIDE + kc]     = *(const v4f*)gl;
      *(v4f*)&lds_blo[n * BSTRIDE + kc + 8] = *(const v4f*)(gl + 8);
    }
    __syncthreads();

    // GEMM1 tile: C(16x16) += A(16x256) * W1^T-slab, bf16 hi/lo compensated.
    // B 32x16 in contiguous k order (matches A): e[j] = B^T[n][32kt + 16g + j].
    // Double-buffered B loads + two independent accumulator chains.
    const __bf16* bbh = &lds_bhi[m * BSTRIDE + 16 * g];
    const __bf16* bbl = &lds_blo[m * BSTRIDE + 16 * g];

    v16bf bh_cur = load_b16(bbh);
    v16bf bl_cur = load_b16(bbl);

    v8f acc0 = {0.f, 0.f, 0.f, 0.f, 0.f, 0.f, 0.f, 0.f};
    v8f acc1 = {0.f, 0.f, 0.f, 0.f, 0.f, 0.f, 0.f, 0.f};
#pragma unroll
    for (int kt = 0; kt < 8; ++kt) {
      v16bf bh_nxt = bh_cur;
      v16bf bl_nxt = bl_cur;
      if (kt < 7) {
        bh_nxt = load_b16(bbh + 32 * (kt + 1));
        bl_nxt = load_b16(bbl + 32 * (kt + 1));
      }
      acc0 = __builtin_amdgcn_wmma_f32_16x16x32_bf16(false, a_hi[kt], false, bh_cur, (short)0, acc0, false, false);
      acc1 = __builtin_amdgcn_wmma_f32_16x16x32_bf16(false, a_lo[kt], false, bh_cur, (short)0, acc1, false, false);
      acc1 = __builtin_amdgcn_wmma_f32_16x16x32_bf16(false, a_hi[kt], false, bl_cur, (short)0, acc1, false, false);
      bh_cur = bh_nxt;
      bl_cur = bl_nxt;
    }

    // bias + relu; stage h tile (C layout: lane holds column m, rows 8g+i)
    const float bias   = b1[16 * nt + m];
    const int   cstage = 16 * (nt & 1) + m;
    float* hs = &lds_h[wave][0];
#pragma unroll
    for (int i = 0; i < 8; ++i) {
      float hv = acc0[i] + acc1[i] + bias;
      hv = hv > 0.f ? hv : 0.f;
      hs[(8 * g + i) * HSTRIDE + cstage] = hv;
    }

    if (nt & 1) {
      // in-wave LDS ordering: writes above must land before transposed reads
      asm volatile("s_wait_dscnt 0" ::: "memory");
      // transpose through LDS: build A2 operand (row = m) for GEMM2 K-tile
      v16bf a2h, a2l;
#pragma unroll
      for (int j = 0; j < 8; ++j) {
        __bf16 h, l;
        bf16_split(hs[m * HSTRIDE + 16 * g + j], h, l);
        a2h[j] = h; a2l[j] = l;
        bf16_split(hs[m * HSTRIDE + 16 * g + 8 + j], h, l);
        a2h[8 + j] = h; a2l[8 + j] = l;
      }
      const int k2t = nt >> 1;
      v16bf b2h = load_b16(&lds_b2hi[m * BSTRIDE + 32 * k2t + 16 * g]);
      v16bf b2l = load_b16(&lds_b2lo[m * BSTRIDE + 32 * k2t + 16 * g]);
      acc2a = __builtin_amdgcn_wmma_f32_16x16x32_bf16(false, a2h, false, b2h, (short)0, acc2a, false, false);
      acc2b = __builtin_amdgcn_wmma_f32_16x16x32_bf16(false, a2l, false, b2h, (short)0, acc2b, false, false);
      acc2b = __builtin_amdgcn_wmma_f32_16x16x32_bf16(false, a2h, false, b2l, (short)0, acc2b, false, false);
    }
  }

  // scores: C layout -> lane holds channel column m (valid for m < CH), rows 8g+i
  if (valid && m < CH) {
    const float bias2 = b2[m];
#pragma unroll
    for (int i = 0; i < 8; ++i) {
      const int mr = 8 * g + i;
      const int e  = tile * 16 + mr;
      const float s = acc2a[i] + acc2b[i] + bias2;
      score_out[(long)e * CH + m] = s;
      const int rr = ei[e];
      atomicMax(&seg_key[rr * CH + m], flip_f(s));
    }
  }
}

// ---- softmax pass 1: e = exp(score - segmax); accumulate segment sums ------
__global__ void softmax_pass1(const int* __restrict__ ei,
                              const unsigned* __restrict__ seg_key,
                              float* __restrict__ seg_sum,
                              float* __restrict__ out, int E) {
  int tid = blockIdx.x * blockDim.x + threadIdx.x;
  if (tid >= E * CH) return;
  int e = tid >> 3, ch = tid & 7;
  int r = ei[e];
  float mx = unflip_f(seg_key[r * CH + ch]);
  float v = __expf(out[tid] - mx);
  out[tid] = v;
  atomicAdd(&seg_sum[r * CH + ch], v);
}

// ---- softmax pass 2: normalize ---------------------------------------------
__global__ void softmax_pass2(const int* __restrict__ ei,
                              const float* __restrict__ seg_sum,
                              float* __restrict__ out, int E) {
  int tid = blockIdx.x * blockDim.x + threadIdx.x;
  if (tid >= E * CH) return;
  int e = tid >> 3, ch = tid & 7;
  out[tid] = out[tid] / (seg_sum[ei[e] * CH + ch] + 1e-16f);
}

// ---------------------------------------------------------------------------
extern "C" void kernel_launch(void* const* d_in, const int* in_sizes, int n_in,
                              void* d_out, int out_size, void* d_ws, size_t ws_size,
                              hipStream_t stream) {
  const float* x  = (const float*)d_in[0];
  const int*   ei = (const int*)d_in[1];
  const float* W1 = (const float*)d_in[2];
  const float* b1 = (const float*)d_in[3];
  const float* W2 = (const float*)d_in[4];
  const float* b2 = (const float*)d_in[5];

  const int E = in_sizes[1] / 2;          // 1,000,000
  const int N = in_sizes[0] / NFEATS;     // 100,000

  // workspace layout (~6.8 MB total)
  char* ws = (char*)d_ws;
  __bf16*   w1t_hi  = (__bf16*)ws;                              // 128 KB
  __bf16*   w1t_lo  = w1t_hi + KDIM * HIDDEN;                   // 128 KB
  __bf16*   w2t_hi  = w1t_lo + KDIM * HIDDEN;                   // 8 KB
  __bf16*   w2t_lo  = w2t_hi + 16 * KDIM;                       // 8 KB
  unsigned* seg_key = (unsigned*)(w2t_lo + 16 * KDIM);          // N*CH*4
  float*    seg_sum = (float*)(seg_key + (size_t)N * CH);       // N*CH*4

  float* out = (float*)d_out;

  prep_w1t<<<(KDIM * HIDDEN) / 256, 256, 0, stream>>>(W1, w1t_hi, w1t_lo);
  prep_w2t<<<(16 * KDIM) / 256, 256, 0, stream>>>(W2, w2t_hi, w2t_lo);

  const int segTotal = N * CH;
  init_seg<<<(segTotal + 255) / 256, 256, 0, stream>>>(seg_key, seg_sum, segTotal);

  const int ntiles  = (E + 15) / 16;
  const int nblocks = (ntiles + 7) / 8;
  edge_mlp_kernel<<<nblocks, 256, 0, stream>>>(x, ei, w1t_hi, w1t_lo,
                                               w2t_hi, w2t_lo, b1, b2,
                                               out, seg_key, E);

  const int tot = E * CH;
  softmax_pass1<<<(tot + 255) / 256, 256, 0, stream>>>(ei, seg_key, seg_sum, out, E);
  softmax_pass2<<<(tot + 255) / 256, 256, 0, stream>>>(ei, seg_sum, out, E);
}
